// Attention_53025666236638
// MI455X (gfx1250) — 16-bit and 8-bit matrix operand order repaired and checked by reading, compiled, not run on hardware
//
#include <hip/hip_runtime.h>
#include <stdint.h>

typedef __attribute__((ext_vector_type(16))) _Float16 v16h;
typedef __attribute__((ext_vector_type(8)))  float    v8f;
typedef __attribute__((ext_vector_type(8)))  int      v8i;
typedef __attribute__((ext_vector_type(4)))  int      v4i;
typedef __attribute__((ext_vector_type(2)))  int      v2i;

#define NN 1024
#define CC 512
#define HH 8
#define HD 64
#define D3 1536

__device__ __forceinline__ float spikef(float x) {
  return floorf(fminf(fmaxf(x, 0.0f), 4.0f) + 0.5f);
}

__device__ __forceinline__ v8i load_a_u8(const uint8_t* row, int half) {
  v4i t0 = *(const v4i*)(row +      16 * half);
  v4i t1 = *(const v4i*)(row + 32 + 16 * half);
  v8i a;
  a[0] = t0[0]; a[1] = t0[1]; a[2] = t0[2]; a[3] = t0[3];
  a[4] = t1[0]; a[5] = t1[1]; a[6] = t1[2]; a[7] = t1[3];
  return a;
}
__device__ __forceinline__ v8i load_b_u8(const uint8_t* colrow, int half) {
  v4i u0 = *(const v4i*)(colrow +      16 * half);
  v4i u1 = *(const v4i*)(colrow + 32 + 16 * half);
  v8i b;
  b[0] = u0[0]; b[1] = u0[1]; b[2] = u0[2]; b[3] = u0[3];
  b[4] = u1[0]; b[5] = u1[1]; b[6] = u1[2]; b[7] = u1[3];
  return b;
}

__global__ __launch_bounds__(256) void k_qkv(const float* __restrict__ x,
                                             const float* __restrict__ Wqkv,
                                             float* __restrict__ qkv) {
  int wid  = blockIdx.x * 8 + (threadIdx.x >> 5);
  int lane = threadIdx.x & 31;
  int col = lane & 15, half = lane >> 4;
  int mt = wid / 24, nt = wid - mt * 24;
  int row0 = mt * 16, col0 = nt * 64;
  const float* xr = x + (size_t)(row0 + col) * CC;
  v8f acc[4] = {};
  for (int k0 = 0; k0 < CC; k0 += 32) {
    v16h A;
#pragma unroll
    for (int v = 0; v < 8; ++v) {
      int kk = (v < 4) ? (8 * half + 2 * v) : (16 + 8 * half + 2 * (v - 4));
      A[2 * v]     = (_Float16)spikef(xr[k0 + kk]);
      A[2 * v + 1] = (_Float16)spikef(xr[k0 + kk + 1]);
    }
#pragma unroll
    for (int t = 0; t < 4; ++t) {
      const float* wr = Wqkv + (size_t)(col0 + t * 16 + col) * CC + k0 + 8 * half;
      v16h Bf;
#pragma unroll
      for (int v = 0; v < 8; ++v) {
        Bf[2 * v]     = (_Float16)wr[2 * v + (v < 4 ? 0 : 8)];
        Bf[2 * v + 1] = (_Float16)wr[2 * v + 1 + (v < 4 ? 0 : 8)];
      }
      acc[t] = __builtin_amdgcn_wmma_f32_16x16x32_f16(false, A, false, Bf,
                                                      (short)0, acc[t], false, false);
    }
  }
#pragma unroll
  for (int t = 0; t < 4; ++t)
#pragma unroll
    for (int r = 0; r < 8; ++r)
      qkv[(size_t)(row0 + r + 8 * half) * D3 + col0 + t * 16 + col] = acc[t][r];
}

__global__ __launch_bounds__(256) void k_bnstat_partial(const float* __restrict__ qkv,
                                                        float* __restrict__ psum,
                                                        float* __restrict__ psumsq) {
  int d = blockIdx.x * 256 + threadIdx.x;
  int part = blockIdx.y;
  float s = 0.f, sq = 0.f;
  for (int r = 0; r < 128; ++r) {
    float v = qkv[(size_t)(part * 128 + r) * D3 + d];
    s += v; sq += v * v;
  }
  psum[part * D3 + d]   = s;
  psumsq[part * D3 + d] = sq;
}

__global__ __launch_bounds__(256) void k_bnstat_final(const float* __restrict__ psum,
                                                      const float* __restrict__ psumsq,
                                                      const float* __restrict__ gamma,
                                                      const float* __restrict__ beta,
                                                      float* __restrict__ scl,
                                                      float* __restrict__ shf) {
  int d = blockIdx.x * 256 + threadIdx.x;
  float s = 0.f, sq = 0.f;
  for (int p = 0; p < 32; ++p) { s += psum[p * D3 + d]; sq += psumsq[p * D3 + d]; }
  float mu   = s * (1.0f / 4096.0f);
  float var  = sq * (1.0f / 4096.0f) - mu * mu;
  float rstd = rsqrtf(var + 1e-5f);
  float g = gamma[d] * rstd;
  scl[d] = g;
  shf[d] = beta[d] - mu * g;
}

__global__ __launch_bounds__(256) void k_norm_spike(const float* __restrict__ qkv,
                                                    const float* __restrict__ scl,
                                                    const float* __restrict__ shf,
                                                    uint8_t* __restrict__ q,
                                                    uint8_t* __restrict__ k,
                                                    uint8_t* __restrict__ vT) {
  size_t idx = (size_t)blockIdx.x * 256 + threadIdx.x;
  int d = (int)(idx % D3);
  int rown = (int)(idx / D3);
  int b = rown >> 10, n = rown & 1023;
  float y = qkv[idx] * scl[d] + shf[d];
  uint8_t s = (uint8_t)(int)spikef(y);
  int which = d / CC, c = d - which * CC;
  int h = c >> 6, dd = c & 63;
  int bh = b * HH + h;
  if (which == 0)      q [(size_t)(bh * NN + n) * HD + dd] = s;
  else if (which == 1) k [(size_t)(bh * NN + n) * HD + dd] = s;
  else                 vT[(size_t)(bh * HD + dd) * NN + n] = s;
}

__global__ __launch_bounds__(256) void k_attn(const uint8_t* __restrict__ q,
                                              const uint8_t* __restrict__ k,
                                              uint8_t* __restrict__ attn) {
  int wid  = blockIdx.x * 8 + (threadIdx.x >> 5);
  int lane = threadIdx.x & 31;
  int col = lane & 15, half = lane >> 4;
  int bh = wid >> 10;
  int rem = wid & 1023;
  int nt = rem >> 4, mt = rem & 15;
  int n0 = nt * 16, m0 = mt * 64;
  const uint8_t* qb = q + (size_t)bh * NN * HD;
  const uint8_t* kb = k + (size_t)bh * NN * HD;
  uint8_t* ab = attn + (size_t)bh * NN * NN;
  v8i A = load_a_u8(qb + (size_t)(n0 + col) * HD, half);
  v8i acc[4];
#pragma unroll
  for (int t = 0; t < 4; ++t) {
    v8i Bf = load_b_u8(kb + (size_t)(m0 + t * 16 + col) * HD, half);
    v8i z = {};
    acc[t] = __builtin_amdgcn_wmma_i32_16x16x64_iu8(false, A, false, Bf, z, false, false);
  }
#pragma unroll
  for (int t = 0; t < 4; ++t)
#pragma unroll
    for (int r = 0; r < 8; ++r) {
      int val = acc[t][r];
      val = val > 4 ? 4 : val;
      ab[(size_t)(n0 + r + 8 * half) * NN + m0 + t * 16 + col] = (uint8_t)val;
    }
}

__global__ __launch_bounds__(256) void k_av(const uint8_t* __restrict__ attn,
                                            const uint8_t* __restrict__ vT,
                                            uint8_t* __restrict__ outs) {
  int wid  = blockIdx.x * 8 + (threadIdx.x >> 5);
  int lane = threadIdx.x & 31;
  int col = lane & 15, half = lane >> 4;
  int bh = wid >> 6;
  int nt = wid & 63;
  int n0 = nt * 16;
  int b = bh >> 3, h = bh & 7;
  const uint8_t* ab = attn + (size_t)bh * NN * NN + (size_t)(n0 + col) * NN;
  const uint8_t* vb = vT + (size_t)bh * HD * NN;
  v8i acc[4] = {};
  for (int kt = 0; kt < 16; ++kt) {
    v8i A = load_a_u8(ab + kt * 64, half);
#pragma unroll
    for (int t = 0; t < 4; ++t) {
      v8i Bf = load_b_u8(vb + (size_t)(t * 16 + col) * NN + kt * 64, half);
      acc[t] = __builtin_amdgcn_wmma_i32_16x16x64_iu8(false, A, false, Bf, acc[t], false, false);
    }
  }
#pragma unroll
  for (int t = 0; t < 4; ++t)
#pragma unroll
    for (int r = 0; r < 8; ++r) {
      float val = (float)acc[t][r] * 0.125f;
      uint8_t s = (uint8_t)(int)spikef(val);
      int n = n0 + r + 8 * half;
      int c = h * HD + t * 16 + col;
      outs[(size_t)(b * NN + n) * CC + c] = s;
    }
}

__global__ __launch_bounds__(256) void k_proj(const uint8_t* __restrict__ outs,
                                              const float* __restrict__ Wproj,
                                              const float* __restrict__ bproj,
                                              float* __restrict__ out) {
  int wid  = blockIdx.x * 8 + (threadIdx.x >> 5);
  int lane = threadIdx.x & 31;
  int col = lane & 15, half = lane >> 4;
  int mt = wid >> 3, nt = wid & 7;
  int row0 = mt * 16, col0 = nt * 64;
  const uint8_t* ar = outs + (size_t)(row0 + col) * CC;
  v8f acc[4] = {};
  for (int k0 = 0; k0 < CC; k0 += 32) {
    v16h A;
#pragma unroll
    for (int v = 0; v < 8; ++v) {
      int kk = (v < 4) ? (8 * half + 2 * v) : (16 + 8 * half + 2 * (v - 4));
      A[2 * v]     = (_Float16)(float)ar[k0 + kk];
      A[2 * v + 1] = (_Float16)(float)ar[k0 + kk + 1];
    }
#pragma unroll
    for (int t = 0; t < 4; ++t) {
      const float* wr = Wproj + (size_t)(col0 + t * 16 + col) * CC + k0 + 8 * half;
      v16h Bf;
#pragma unroll
      for (int v = 0; v < 8; ++v) {
        Bf[2 * v]     = (_Float16)wr[2 * v + (v < 4 ? 0 : 8)];
        Bf[2 * v + 1] = (_Float16)wr[2 * v + 1 + (v < 4 ? 0 : 8)];
      }
      acc[t] = __builtin_amdgcn_wmma_f32_16x16x32_f16(false, A, false, Bf,
                                                      (short)0, acc[t], false, false);
    }
  }
#pragma unroll
  for (int t = 0; t < 4; ++t) {
    int e = col0 + t * 16 + col;
    float bias = bproj[e];
#pragma unroll
    for (int r = 0; r < 8; ++r)
      out[(size_t)(row0 + r + 8 * half) * CC + e] = acc[t][r] + bias;
  }
}

extern "C" void kernel_launch(void* const* d_in, const int* in_sizes, int n_in,
                              void* d_out, int out_size, void* d_ws, size_t ws_size,
                              hipStream_t stream) {
  const float* x     = (const float*)d_in[0];
  const float* Wqkv  = (const float*)d_in[1];
  const float* gamma = (const float*)d_in[2];
  const float* beta  = (const float*)d_in[3];
  const float* Wproj = (const float*)d_in[4];
  const float* bproj = (const float*)d_in[5];
  float* out = (float*)d_out;

  uint8_t* ws = (uint8_t*)d_ws;
  size_t off = 0;
  auto alloc = [&](size_t bytes) -> void* {
    void* p = ws + off;
    off = (off + bytes + 255) & ~(size_t)255;
    return p;
  };
  float*   qkv    = (float*)alloc((size_t)4096 * D3 * 4);
  float*   psum   = (float*)alloc((size_t)32 * D3 * 4);
  float*   psumsq = (float*)alloc((size_t)32 * D3 * 4);
  float*   scl    = (float*)alloc((size_t)D3 * 4);
  float*   shf    = (float*)alloc((size_t)D3 * 4);
  uint8_t* q      = (uint8_t*)alloc((size_t)32 * NN * HD);
  uint8_t* k      = (uint8_t*)alloc((size_t)32 * NN * HD);
  uint8_t* vT     = (uint8_t*)alloc((size_t)32 * NN * HD);
  uint8_t* attn   = (uint8_t*)alloc((size_t)32 * NN * NN);
  uint8_t* outs   = (uint8_t*)alloc((size_t)4096 * CC);

  k_qkv          <<<768, 256, 0, stream>>>(x, Wqkv, qkv);
  k_bnstat_partial<<<dim3(6, 32), 256, 0, stream>>>(qkv, psum, psumsq);
  k_bnstat_final <<<6, 256, 0, stream>>>(psum, psumsq, gamma, beta, scl, shf);
  k_norm_spike   <<<24576, 256, 0, stream>>>(qkv, scl, shf, q, k, vT);
  k_attn         <<<4096, 256, 0, stream>>>(q, k, attn);
  k_av           <<<256, 256, 0, stream>>>(attn, vT, outs);
  k_proj         <<<256, 256, 0, stream>>>(outs, Wproj, bproj, out);
}
